// PSWarpHead_27315992002666
// MI455X (gfx1250) — compile-verified, matrix operand order checked by reading, not run on hardware
//
#include <hip/hip_runtime.h>
#include <hip/hip_bf16.h>

typedef __bf16 v16bf __attribute__((ext_vector_type(16)));
typedef float  v8f   __attribute__((ext_vector_type(8)));
typedef unsigned int u32x4 __attribute__((ext_vector_type(4)));
typedef unsigned short u16;

#define H_IMG 200
#define W_IMG 176
#define CIN   256
#define PCH   28
#define NBOX  2048
#define NB    4

union FragU { u32x4 q[2]; v16bf v; };

__device__ __forceinline__ u16 f2bf(float f) {
  unsigned u = __float_as_uint(f);
  unsigned r = u + 0x7fffu + ((u >> 16) & 1u);   // round-to-nearest-even
  return (u16)(r >> 16);
}

// ---------------------------------------------------------------------------
// Kernel 1: repack weights into WMMA B-fragment layout (bf16) + fold BN.
// pw1 frag index: frag = (tap*8 + cb)*2 + nt ; per frag: 32 lanes x 16 bf16.
// Element e of lane's v16bf maps to K = (e<8 ? e : e+8) + (lane<16 ? 0 : 8),
// N = lane%16  (mirrors the 16-bit A layout on the K axis).
// ---------------------------------------------------------------------------
__global__ void prepack_kernel(const float* __restrict__ w1,
                               const float* __restrict__ w2,
                               const float* __restrict__ gamma,
                               const float* __restrict__ beta,
                               const float* __restrict__ mean,
                               const float* __restrict__ var,
                               u16* __restrict__ pw1,
                               u16* __restrict__ pw2,
                               float* __restrict__ bnsc,
                               float* __restrict__ bnsh) {
  const int NW1 = 9 * 8 * 2 * 32 * 16;   // 73728
  int i = blockIdx.x * blockDim.x + threadIdx.x;
  if (i < NW1) {
    int e    = i & 15;
    int lane = (i >> 4) & 31;
    int frag = i >> 9;
    int nt   = frag & 1;
    int cb   = (frag >> 1) & 7;
    int tap  = frag >> 4;
    int kk = (e < 8) ? e : e + 8;
    int K  = kk + ((lane < 16) ? 0 : 8);
    int N  = lane & 15;
    int ci = cb * 32 + K;
    int co = nt * 16 + N;
    int ky = tap / 3, kx = tap % 3;
    float v = 0.f;
    if (co < PCH) v = w1[(((size_t)co * CIN + ci) * 3 + ky) * 3 + kx];
    pw1[i] = f2bf(v);
  } else if (i < NW1 + 1024) {
    int j = i - NW1;
    int e = j & 15, lane = (j >> 4) & 31, nt = j >> 9;
    int kk = (e < 8) ? e : e + 8;
    int K  = kk + ((lane < 16) ? 0 : 8);
    int N  = lane & 15;
    int co = nt * 16 + N;
    float v = 0.f;
    if (co < PCH && K < PCH) v = w2[co * PCH + K];
    pw2[j] = f2bf(v);
  } else if (i < NW1 + 1024 + 32) {
    int c = i - NW1 - 1024;
    float s = 0.f, sh = 0.f;
    if (c < PCH) {
      float inv = gamma[c] * rsqrtf(var[c] + 0.001f);
      s  = inv;
      sh = beta[c] - mean[c] * inv;
    }
    bnsc[c] = s;
    bnsh[c] = sh;
  }
}

// ---------------------------------------------------------------------------
// Kernel 2: fused conv3x3 (bf16 WMMA implicit GEMM) + BN + ReLU + conv1x1.
// One workgroup = (batch b, output rows y0..y0+1). 8 waves x 3 M-tiles of 16
// columns (24 tiles = 2 rows x 12 padded tiles). K = 9 taps x 8 ci-blocks.
// ---------------------------------------------------------------------------
__global__ __launch_bounds__(256) void conv_fused_kernel(
    const float* __restrict__ x,
    const u16*   __restrict__ pw1,
    const u16*   __restrict__ pw2,
    const float* __restrict__ bnsc,
    const float* __restrict__ bnsh,
    float* __restrict__ feat2) {
  __shared__ u16 smem[4 * 180 * 32];   // 46080 B window (aliased later as bounce)

  const int blk  = blockIdx.x;
  const int b    = blk / 100;
  const int y0   = (blk % 100) * 2;
  const int tid  = threadIdx.x;
  const int lane = tid & 31;
  const int wave = tid >> 5;

  v8f acc[3][2];
#pragma unroll
  for (int i = 0; i < 3; ++i) { acc[i][0] = 0.f; acc[i][1] = 0.f; }

  const u32x4* pq1 = (const u32x4*)pw1;

  for (int cb = 0; cb < 8; ++cb) {
    __syncthreads();
    // Stage 4-row x 180-col x 32-ci window into LDS as bf16, ci-contiguous.
    const float* xb = x + ((size_t)b * CIN + (size_t)cb * 32) * (H_IMG * W_IMG);
    for (int i = tid; i < 4 * 180 * 32; i += 256) {
      int cl   = i % 180;
      int rest = i / 180;
      int ci   = rest & 31;
      int row  = rest >> 5;
      int yy = y0 - 1 + row;
      int xx = cl - 1;
      float v = 0.f;
      if (yy >= 0 && yy < H_IMG && xx >= 0 && xx < W_IMG)
        v = xb[(size_t)ci * (H_IMG * W_IMG) + yy * W_IMG + xx];
      smem[(row * 180 + cl) * 32 + ci] = f2bf(v);
    }
    __syncthreads();

    for (int tap = 0; tap < 9; ++tap) {
      const int ky = tap / 3, kx = tap % 3;
      // B fragments (shared across the wave's 3 M-tiles), streamed from L2.
      FragU bf0, bf1;
      int fb = (tap * 8 + cb) * 2;
      bf0.q[0] = pq1[((fb + 0) * 32 + lane) * 2 + 0];
      bf0.q[1] = pq1[((fb + 0) * 32 + lane) * 2 + 1];
      bf1.q[0] = pq1[((fb + 1) * 32 + lane) * 2 + 0];
      bf1.q[1] = pq1[((fb + 1) * 32 + lane) * 2 + 1];
#pragma unroll
      for (int i = 0; i < 3; ++i) {
        int t    = wave + 8 * i;
        int rowt = t / 12, xt = t % 12;
        int m    = lane & 15;
        int cl   = xt * 16 + m + kx;
        int row  = rowt + ky;
        int cib  = (lane < 16) ? 0 : 8;
        const u16* p = &smem[(row * 180 + cl) * 32];
        FragU af;
        af.q[0] = *(const u32x4*)(p + cib);
        af.q[1] = *(const u32x4*)(p + cib + 16);
        acc[i][0] = __builtin_amdgcn_wmma_f32_16x16x32_bf16(
            false, af.v, false, bf0.v, (short)0, acc[i][0], false, false);
        acc[i][1] = __builtin_amdgcn_wmma_f32_16x16x32_bf16(
            false, af.v, false, bf1.v, (short)0, acc[i][1], false, false);
      }
    }
  }

  // ---------------- Epilogue: BN + ReLU -> LDS bounce (C-layout -> A-layout)
  __syncthreads();
  u16* bounce = smem;   // reuse window LDS; per-wave region 3 x (16x32) bf16
#pragma unroll
  for (int nt = 0; nt < 2; ++nt) {
    int c = (lane & 15) + nt * 16;
    float sc = bnsc[c];
    float sh = bnsh[c];
#pragma unroll
    for (int i = 0; i < 3; ++i) {
#pragma unroll
      for (int r = 0; r < 8; ++r) {
        int M = r + ((lane < 16) ? 0 : 8);
        float v = fmaxf(acc[i][nt][r] * sc + sh, 0.f);
        bounce[(wave * 3 + i) * 512 + M * 32 + c] = f2bf(v);
      }
    }
  }
  __syncthreads();

  // conv 1x1 (K=28 padded to 32) with prepacked B fragments.
  const u32x4* pq2 = (const u32x4*)pw2;
  FragU b20, b21;
  b20.q[0] = pq2[(0 * 32 + lane) * 2 + 0];
  b20.q[1] = pq2[(0 * 32 + lane) * 2 + 1];
  b21.q[0] = pq2[(1 * 32 + lane) * 2 + 0];
  b21.q[1] = pq2[(1 * 32 + lane) * 2 + 1];

#pragma unroll
  for (int i = 0; i < 3; ++i) {
    int t    = wave + 8 * i;
    int rowt = t / 12, xt = t % 12;
    const u16* p = &bounce[(wave * 3 + i) * 512];
    int mm  = lane & 15;
    int cib = (lane < 16) ? 0 : 8;
    FragU af;
    af.q[0] = *(const u32x4*)(p + mm * 32 + cib);
    af.q[1] = *(const u32x4*)(p + mm * 32 + cib + 16);
    v8f d0 = 0.f, d1 = 0.f;
    d0 = __builtin_amdgcn_wmma_f32_16x16x32_bf16(false, af.v, false, b20.v,
                                                 (short)0, d0, false, false);
    d1 = __builtin_amdgcn_wmma_f32_16x16x32_bf16(false, af.v, false, b21.v,
                                                 (short)0, d1, false, false);
    int y = y0 + rowt;
#pragma unroll
    for (int nt = 0; nt < 2; ++nt) {
      v8f d = nt ? d1 : d0;
      int c = (lane & 15) + nt * 16;
      if (c < PCH) {
#pragma unroll
        for (int r = 0; r < 8; ++r) {
          int M  = r + ((lane < 16) ? 0 : 8);
          int xp = xt * 16 + M;
          if (xp < W_IMG)
            feat2[(((size_t)b * PCH + c) * H_IMG + y) * W_IMG + xp] = d[r];
        }
      }
    }
  }
}

// ---------------------------------------------------------------------------
// Kernel 3: rotated-window bilinear sampling + mean over 28 points.
// ---------------------------------------------------------------------------
__device__ __forceinline__ float gather0(const float* img, int yi, int xi) {
  if (xi < 0 || xi > W_IMG - 1 || yi < 0 || yi > H_IMG - 1) return 0.f;
  return img[yi * W_IMG + xi];
}

__global__ void sample_kernel(const float* __restrict__ boxes,
                              const float* __restrict__ feat2,
                              float* __restrict__ out) {
  int idx = blockIdx.x * blockDim.x + threadIdx.x;
  if (idx >= NB * NBOX) return;
  int b = idx >> 11;
  const float* bx = boxes + (size_t)idx * 7;
  float xg = bx[0], yg = bx[1], wg = bx[3], lg = bx[4], rg = bx[6];
  float cosT = __cosf(rg), sinT = __sinf(rg);
  const float* fb = feat2 + (size_t)b * PCH * H_IMG * W_IMG;
  float accum = 0.f;
#pragma unroll
  for (int p = 0; p < PCH; ++p) {
    int ii = p / 7, jj = p % 7;
    float lx = (float)ii * (1.f / 3.f) - 0.5f;   // linspace(-0.5,0.5,4)
    float ly = (float)jj * (1.f / 6.f) - 0.5f;   // linspace(-0.5,0.5,7)
    float xx = lx * wg, yy = ly * lg;
    float gx = (xx * cosT + yy * sinT + xg) * 2.5f;          // +GRID_OFF.x(0)
    float gy = ((yy * cosT - xx * sinT + yg) + 40.f) * 2.5f; // +GRID_OFF.y
    const float* img = fb + (size_t)p * (H_IMG * W_IMG);
    float x0f = floorf(gx), y0f = floorf(gy);
    int x0 = (int)x0f, y0i = (int)y0f;
    float fx = gx - x0f, fy = gy - y0f;
    float Ia = gather0(img, y0i,     x0);
    float Ib = gather0(img, y0i + 1, x0);
    float Ic = gather0(img, y0i,     x0 + 1);
    float Id = gather0(img, y0i + 1, x0 + 1);
    accum += (1.f - fx) * (1.f - fy) * Ia + (1.f - fx) * fy * Ib +
             fx * (1.f - fy) * Ic + fx * fy * Id;
  }
  out[idx] = accum * (1.f / 28.f);
}

// ---------------------------------------------------------------------------
extern "C" void kernel_launch(void* const* d_in, const int* in_sizes, int n_in,
                              void* d_out, int out_size, void* d_ws,
                              size_t ws_size, hipStream_t stream) {
  const float* x     = (const float*)d_in[0];
  const float* boxes = (const float*)d_in[1];
  const float* w1    = (const float*)d_in[2];
  const float* gamma = (const float*)d_in[3];
  const float* beta  = (const float*)d_in[4];
  const float* mean  = (const float*)d_in[5];
  const float* var   = (const float*)d_in[6];
  const float* w2    = (const float*)d_in[7];

  char* ws = (char*)d_ws;
  const size_t FEAT2_B = (size_t)NB * PCH * H_IMG * W_IMG * sizeof(float); // 15,769,600
  float* feat2 = (float*)ws;
  u16*   pw1   = (u16*)(ws + FEAT2_B);
  u16*   pw2   = (u16*)(ws + FEAT2_B + 147456);
  float* bnsc  = (float*)(ws + FEAT2_B + 147456 + 2048);
  float* bnsh  = bnsc + 32;

  const int NPACK = 9 * 8 * 2 * 32 * 16 + 1024 + 64;
  prepack_kernel<<<(NPACK + 255) / 256, 256, 0, stream>>>(
      w1, w2, gamma, beta, mean, var, pw1, pw2, bnsc, bnsh);

  conv_fused_kernel<<<NB * 100, 256, 0, stream>>>(x, pw1, pw2, bnsc, bnsh,
                                                  feat2);

  sample_kernel<<<(NB * NBOX + 255) / 256, 256, 0, stream>>>(boxes, feat2,
                                                             (float*)d_out);
}
